// RNNEncoder_86930138071297
// MI455X (gfx1250) — compile-verified, matrix operand order checked by reading, not run on hardware
//
#include <hip/hip_runtime.h>
#include <hip/hip_bf16.h>
#include <stdint.h>

// Problem sizes (fixed by the reference)
#define B_   64
#define T_   1024
#define E_   256
#define U_   1024
#define N3U  3072   // 3*U

typedef __attribute__((ext_vector_type(16))) __bf16 v16bf;
typedef __attribute__((ext_vector_type(8)))  float  v8f;

static __device__ __forceinline__ v8f vzero() {
  v8f v;
#pragma unroll
  for (int i = 0; i < 8; ++i) v[i] = 0.f;
  return v;
}

// Build a CDNA5 16-bit A-matrix 16x32 fragment from an fp32 row.
// Lane L<16 holds row M=L, K in {0..7, 16..23}; lane L>=16 holds row M=L-16,
// K in {8..15, 24..31}.  Caller passes the row pointer for its M and hi=lane>>4.
static __device__ __forceinline__ v16bf load_a16x32_f32(const float* row, int kt, int hi) {
  const float4* p = (const float4*)(row + kt * 32 + hi * 8);
  float4 f0 = p[0];   // K+0..3
  float4 f1 = p[1];   // K+4..7
  float4 f2 = p[4];   // K+16..19
  float4 f3 = p[5];   // K+20..23
  v16bf a;
  a[0]  = (__bf16)f0.x; a[1]  = (__bf16)f0.y; a[2]  = (__bf16)f0.z; a[3]  = (__bf16)f0.w;
  a[4]  = (__bf16)f1.x; a[5]  = (__bf16)f1.y; a[6]  = (__bf16)f1.z; a[7]  = (__bf16)f1.w;
  a[8]  = (__bf16)f2.x; a[9]  = (__bf16)f2.y; a[10] = (__bf16)f2.z; a[11] = (__bf16)f2.w;
  a[12] = (__bf16)f3.x; a[13] = (__bf16)f3.y; a[14] = (__bf16)f3.z; a[15] = (__bf16)f3.w;
  return a;
}

static __device__ __forceinline__ v16bf load_b_q2(const uint4* q) {
  union { uint4 u[2]; v16bf v; } t;
  t.u[0] = q[0];
  t.u[1] = q[1];
  return t.v;
}

#define WMMA_BF16(a, b, c) \
  __builtin_amdgcn_wmma_f32_16x16x32_bf16(false, (a), false, (b), (short)0, (c), false, false)

// ---------------------------------------------------------------------------
// Pre-swizzle an fp32 [K x N] row-major weight matrix into the CDNA5 16-bit
// B-matrix (KxN per 32x16 tile) per-lane layout, stored bf16.
// Storage: tile (nt, kt), lane l -> 16 bf16 (32B) at
//   out[ ((nt*KT + kt)*32 + l) * 16 ]
// Lane content: n = nt*16 + (l&15), kb = kt*32 + (l>>4)*8,
//   half[j] = W[kb+j][n], half[8+j] = W[kb+16+j][n]
// ---------------------------------------------------------------------------
__global__ void k_swizzleW(const float* __restrict__ W, unsigned short* __restrict__ out,
                           int K, int N) {
  int  KT    = K / 32;
  long gid   = (long)blockIdx.x * blockDim.x + threadIdx.x;
  long total = (long)(N / 16) * KT * 32;
  if (gid >= total) return;
  int  l  = (int)(gid & 31);
  long r  = gid >> 5;
  int  kt = (int)(r % KT);
  int  nt = (int)(r / KT);
  int  n  = nt * 16 + (l & 15);
  int  kb = kt * 32 + (l >> 4) * 8;
  union { unsigned short h[16]; uint4 q[2]; } u;
#pragma unroll
  for (int j = 0; j < 8; ++j) {
    u.h[j]     = __builtin_bit_cast(unsigned short, (__bf16)W[(long)(kb + j) * N + n]);
    u.h[8 + j] = __builtin_bit_cast(unsigned short, (__bf16)W[(long)(kb + 16 + j) * N + n]);
  }
  uint4* dst = (uint4*)(out + gid * 16);
  dst[0] = u.q[0];
  dst[1] = u.q[1];
}

// h0 <- initial ; reset grid barrier
__global__ void k_init(const float* __restrict__ initial, float* __restrict__ h0,
                       unsigned* __restrict__ bar) {
  long i = (long)blockIdx.x * blockDim.x + threadIdx.x;
  if (i < (long)B_ * U_) h0[i] = initial[i];
  if (i < 2) bar[i] = 0u;
}

// ---------------------------------------------------------------------------
// Fused embedding gather + input projection:  xp[m, :] = emb[x[m]] @ W_in + b_in
// m = b*T + t in [0, 65536).  Each wave computes a 16(M) x 64(N) tile.
// ---------------------------------------------------------------------------
__global__ __launch_bounds__(256) void k_inproj(const int* __restrict__ x,
                                                const float* __restrict__ emb,
                                                const unsigned short* __restrict__ winS,
                                                const float* __restrict__ b_in,
                                                float* __restrict__ xp) {
  const int lane = threadIdx.x & 31;
  const int wv   = threadIdx.x >> 5;
  const long NT64 = N3U / 64;                       // 48
  long wid  = (long)blockIdx.x * 8 + wv;
  long mt   = wid / NT64;
  int  nt64 = (int)(wid % NT64);
  if (mt >= (long)(B_ * T_) / 16) return;
  const int hi   = lane >> 4;
  const int lrow = lane & 15;

  long mrow = mt * 16 + lrow;
  int  tok  = x[mrow];
  const float* arow = emb + (long)tok * E_;

  v8f acc[4];
#pragma unroll
  for (int j = 0; j < 4; ++j) acc[j] = vzero();

#pragma unroll
  for (int kt = 0; kt < E_ / 32; ++kt) {            // 8 K-steps
    v16bf a = load_a16x32_f32(arow, kt, hi);
#pragma unroll
    for (int j = 0; j < 4; ++j) {
      int ntile = nt64 * 4 + j;
      const uint4* bq = (const uint4*)winS + ((long)(ntile * (E_ / 32) + kt) * 32 + lane) * 2;
      v16bf b = load_b_q2(bq);
      acc[j] = WMMA_BF16(a, b, acc[j]);
    }
  }

#pragma unroll
  for (int j = 0; j < 4; ++j) {
    int   col  = nt64 * 64 + j * 16 + (lane & 15);
    float bias = b_in[col];
#pragma unroll
    for (int v = 0; v < 8; ++v) {
      long m = mt * 16 + v + 8 * hi;
      xp[m * (long)N3U + col] = acc[j][v] + bias;
    }
  }
}

// ---------------------------------------------------------------------------
// Persistent GRU scan.  64 WGs x 128 threads (4 waves).  WG owns 16 hidden
// units (columns n0..n0+15 of h) and stages its 3-gate W_rec slice (96 KB
// bf16, pre-swizzled) in LDS once.  Per step: 3 accumulators x 32 K-tiles of
// v_wmma_f32_16x16x32_bf16, gate math in fp32, double-buffered h, grid
// barrier between steps.
// ---------------------------------------------------------------------------
__global__ __launch_bounds__(128) void k_scan(const float* __restrict__ xp,
                                              const unsigned short* __restrict__ wrecS,
                                              const float* __restrict__ b_rec,
                                              float* h0buf, float* h1buf,
                                              float* __restrict__ out,
                                              unsigned* bar) {
  __shared__ uint4 ldsW[6144];                      // 3 gates * 2048 uint4 = 96 KB

  const int tid  = threadIdx.x;
  const int lane = tid & 31;
  const int wv   = tid >> 5;
  const int blk  = blockIdx.x;                      // 0..63 -> 16 hidden units
  const int n0   = blk * 16;
  const int m0   = wv * 16;

  // Stage the WG's W_rec slice: gate g lives at swizzled ntile (g*64 + blk),
  // each ntile slice is a contiguous 2048-uint4 (32 KB) chunk.
  const uint4* src = (const uint4*)wrecS;
  for (int i = tid; i < 3 * 2048; i += 128) {
    int g = i >> 11;
    int o = i & 2047;
    ldsW[i] = src[(long)(g * 64 + blk) * 2048 + o];
  }
  __syncthreads();

  const int   hi   = lane >> 4;
  const int   lcol = lane & 15;
  const int   n    = n0 + lcol;
  const float brz  = b_rec[n];
  const float brr  = b_rec[n + U_];
  const float brh  = b_rec[n + 2 * U_];

  float* hb[2] = {h0buf, h1buf};

  for (int t = 0; t < T_; ++t) {
    const float* hc = hb[t & 1];
    float*       hn = hb[(t + 1) & 1];

    v8f cz = vzero(), cr = vzero(), chh = vzero();
    const float* arow = hc + (long)(m0 + lcol) * U_;

#pragma unroll 4
    for (int kt = 0; kt < U_ / 32; ++kt) {          // 32 K-steps
      v16bf a  = load_a16x32_f32(arow, kt, hi);
      v16bf bz = load_b_q2(&ldsW[((0 * 32 + kt) * 32 + lane) * 2]);
      v16bf br = load_b_q2(&ldsW[((1 * 32 + kt) * 32 + lane) * 2]);
      v16bf bh = load_b_q2(&ldsW[((2 * 32 + kt) * 32 + lane) * 2]);
      cz  = WMMA_BF16(a, bz, cz);
      cr  = WMMA_BF16(a, br, cr);
      chh = WMMA_BF16(a, bh, chh);
    }

#pragma unroll
    for (int v = 0; v < 8; ++v) {
      int  m     = m0 + v + 8 * hi;
      long xbase = ((long)m * T_ + t) * N3U;
      float xz = xp[xbase + n];
      float xr = xp[xbase + U_ + n];
      float xh = xp[xbase + 2 * U_ + n];
      float rz = cz[v]  + brz;
      float rr = cr[v]  + brr;
      float rh = chh[v] + brh;
      float z  = 1.f / (1.f + __expf(-(xz + rz)));
      float r  = 1.f / (1.f + __expf(-(xr + rr)));
      float hc_ = tanhf(xh + r * rh);
      float hold = hc[(long)m * U_ + n];
      float hv = z * hold + (1.f - z) * hc_;
      hn[(long)m * U_ + n] = hv;
      out[((long)m * T_ + t) * U_ + n] = hv;
      if (t == T_ - 1) out[(long)B_ * T_ * U_ + (long)m * U_ + n] = hv;
    }

    // ---- grid barrier (sense-reversing, 64 WGs) ----
    __syncthreads();
    if (tid == 0) {
      __threadfence();
      unsigned gen = __hip_atomic_load(&bar[1], __ATOMIC_RELAXED, __HIP_MEMORY_SCOPE_AGENT);
      unsigned arrived =
          __hip_atomic_fetch_add(&bar[0], 1u, __ATOMIC_ACQ_REL, __HIP_MEMORY_SCOPE_AGENT);
      if (arrived == 63u) {
        __hip_atomic_store(&bar[0], 0u, __ATOMIC_RELAXED, __HIP_MEMORY_SCOPE_AGENT);
        __hip_atomic_store(&bar[1], gen + 1u, __ATOMIC_RELEASE, __HIP_MEMORY_SCOPE_AGENT);
      } else {
        while (__hip_atomic_load(&bar[1], __ATOMIC_ACQUIRE, __HIP_MEMORY_SCOPE_AGENT) == gen) {
          __builtin_amdgcn_s_sleep(2);
        }
      }
    }
    __syncthreads();
    __threadfence();   // acquire side for all waves before reading new h
  }
}

// ---------------------------------------------------------------------------
extern "C" void kernel_launch(void* const* d_in, const int* in_sizes, int n_in,
                              void* d_out, int out_size, void* d_ws, size_t ws_size,
                              hipStream_t stream) {
  const int*   x       = (const int*)d_in[0];
  const float* initial = (const float*)d_in[1];
  const float* emb     = (const float*)d_in[2];
  const float* W_in    = (const float*)d_in[3];
  const float* W_rec   = (const float*)d_in[4];
  const float* b_in    = (const float*)d_in[5];
  const float* b_rec   = (const float*)d_in[6];
  float*       out     = (float*)d_out;

  char*  ws  = (char*)d_ws;
  size_t off = 0;
  float* xp = (float*)(ws + off);           off += (size_t)B_ * T_ * N3U * 4;   // 805 MB
  unsigned short* winS  = (unsigned short*)(ws + off); off += (size_t)E_ * N3U * 2;
  unsigned short* wrecS = (unsigned short*)(ws + off); off += (size_t)U_ * N3U * 2;
  float* h0 = (float*)(ws + off);           off += (size_t)B_ * U_ * 4;
  float* h1 = (float*)(ws + off);           off += (size_t)B_ * U_ * 4;
  unsigned* bar = (unsigned*)(ws + off);    off += 256;
  (void)ws_size; (void)in_sizes; (void)n_in; (void)out_size;

  // 1) Pre-swizzle weights into CDNA5 bf16 B-matrix lane layout
  {
    long tot = (long)(N3U / 16) * (E_ / 32) * 32;   // 49152
    k_swizzleW<<<(int)((tot + 255) / 256), 256, 0, stream>>>(W_in, winS, E_, N3U);
  }
  {
    long tot = (long)(N3U / 16) * (U_ / 32) * 32;   // 196608
    k_swizzleW<<<(int)((tot + 255) / 256), 256, 0, stream>>>(W_rec, wrecS, U_, N3U);
  }

  // 2) h0 <- initial, reset barrier
  k_init<<<(B_ * U_ + 255) / 256, 256, 0, stream>>>(initial, h0, bar);

  // 3) Fused embedding gather + input projection GEMM (65536 x 3072 x 256)
  k_inproj<<<24576, 256, 0, stream>>>(x, emb, winS, b_in, xp);

  // 4) Persistent GRU scan (1024 steps, grid-synced)
  k_scan<<<64, 128, 0, stream>>>(xp, wrecS, b_rec, h0, h1, out, bar);
}
